// DMN_25331717112464
// MI455X (gfx1250) — compile-verified, matrix operand order checked by reading, not run on hardware
//
#include <hip/hip_runtime.h>
#include <math.h>

// ---------------- constants from the reference ----------------
#define Hh 512
#define Vv 32000
#define Bb 128
#define NFf 40
#define Ll 12
#define QLl 16
#define NDd 4
#define EPISODES 3
#define START_IDX 1

// ---------------- vector types ----------------
typedef __attribute__((ext_vector_type(16))) _Float16 v16h;
typedef __attribute__((ext_vector_type(8)))  _Float16 half8;
typedef __attribute__((ext_vector_type(8)))  float    v8f;
typedef __attribute__((ext_vector_type(4)))  float    f32x4;

union FragU { v16h v; half8 h[2]; };

// ---------------- WMMA GEMM:  C(MxN) = A(MxK) @ W(NxK)^T (+bias)(+relu) ----
// Block tile 128x128, K-step 32, double-buffered LDS, 256 threads = 8 waves in
// 2(M) x 4(N); each wave owns a 64x32 sub-tile = 8 v_wmma_f32_16x16x32_f16
// accumulators fed by 4 A-frags + 2 B-frags per K-step (6 ds_load : 8 wmma).
// Pipeline per K-step: issue next-tile global_load_b128s -> ds_load frags ->
// 8x wmma -> convert+ds_store next tile -> single s_barrier.
// Requires M%128==0, N%128==0, K%32==0 (true for every call in this model).
#define TM 128
#define TN 128
#define TK 32
#define LDS_S (TK + 8)          // 40 halves -> 80B row stride (16B-aligned)
#define ATILE (TM * LDS_S)      // halves per A buffer
#define BTILE (TN * LDS_S)      // halves per B buffer

__global__ __launch_bounds__(256) void wmma_gemm_nt(
    const float* __restrict__ A, const float* __restrict__ W,
    const float* __restrict__ bias, float* __restrict__ C,
    int M, int N, int K, int act)
{
  __shared__ __attribute__((aligned(16))) _Float16 As[2 * ATILE];
  __shared__ __attribute__((aligned(16))) _Float16 Bs[2 * BTILE];

  const int tid  = threadIdx.x;
  const int lane = tid & 31;
  const int wave = tid >> 5;
  const int wm   = wave >> 2;   // 0..1  (64-row strip)
  const int wn   = wave & 3;    // 0..3  (32-col strip)
  const int rowBase = blockIdx.y * TM;
  const int colBase = blockIdx.x * TN;

  // staging: thread (r = tid>>1, hh = tid&1) owns 16 consecutive floats of the
  // 128x32 tile for both A and W (identical shapes).
  const int sr = tid >> 1, shh = tid & 1;
  const float* gA = A + (size_t)(rowBase + sr) * K + shh * 16;
  const float* gW = W + (size_t)(colBase + sr) * K + shh * 16;
  _Float16* sA = As + sr * LDS_S + shh * 16;
  _Float16* sB = Bs + sr * LDS_S + shh * 16;

  f32x4 ar[4], br[4];

  auto gload = [&](int kk) {
    const f32x4* pa = (const f32x4*)(gA + kk);
    const f32x4* pb = (const f32x4*)(gW + kk);
#pragma unroll
    for (int q = 0; q < 4; ++q) { ar[q] = pa[q]; br[q] = pb[q]; }
  };
  auto sstore = [&](int buf) {
    half8 ta0, ta1, tb0, tb1;
#pragma unroll
    for (int q = 0; q < 2; ++q)
#pragma unroll
      for (int e = 0; e < 4; ++e) {
        ta0[4 * q + e] = (_Float16)ar[q][e];
        ta1[4 * q + e] = (_Float16)ar[2 + q][e];
        tb0[4 * q + e] = (_Float16)br[q][e];
        tb1[4 * q + e] = (_Float16)br[2 + q][e];
      }
    half8* da = (half8*)(sA + buf * ATILE);
    half8* db = (half8*)(sB + buf * BTILE);
    da[0] = ta0; da[1] = ta1;
    db[0] = tb0; db[1] = tb1;
  };

  v8f c[4][2];
#pragma unroll
  for (int i = 0; i < 4; ++i)
#pragma unroll
    for (int j = 0; j < 2; ++j) c[i][j] = (v8f){};

  const int half = lane >> 4, fr = lane & 15;
  const _Float16* abase = As + (wm * 64 + fr) * LDS_S + half * 8;
  const _Float16* bbase = Bs + (wn * 32 + fr) * LDS_S + half * 8;

  const int nTiles = K / TK;
  // prologue: stage tile 0 into buffer 0
  gload(0);
  sstore(0);

  for (int tile = 0; tile < nTiles; ++tile) {
    __syncthreads();                       // buffer (tile&1) ready for all waves
    const int cur = tile & 1;
    const bool more = (tile + 1) < nTiles;
    if (more) gload((tile + 1) * TK);      // issue next-tile global loads

    FragU a[4], b[2];
#pragma unroll
    for (int i = 0; i < 4; ++i) {
      const _Float16* p = abase + cur * ATILE + i * 16 * LDS_S;
      a[i].h[0] = *(const half8*)p;
      a[i].h[1] = *(const half8*)(p + 16);
    }
#pragma unroll
    for (int j = 0; j < 2; ++j) {
      const _Float16* p = bbase + cur * BTILE + j * 16 * LDS_S;
      b[j].h[0] = *(const half8*)p;
      b[j].h[1] = *(const half8*)(p + 16);
    }

#pragma unroll
    for (int i = 0; i < 4; ++i) {
#pragma unroll
      for (int j = 0; j < 2; ++j) {
        c[i][j] = __builtin_amdgcn_wmma_f32_16x16x32_f16(
            false, a[i].v, false, b[j].v, (short)0, c[i][j], false, false);
      }
    }

    if (more) sstore(cur ^ 1);             // convert + store into other buffer
  }

  // --- epilogue: 16x16 f32 C layout: lane<16 -> (M=r, N=lane); lane>=16 -> M=8+r
  const int l = lane & 15, hi = lane >> 4;
#pragma unroll
  for (int j = 0; j < 2; ++j) {
    const int col = colBase + wn * 32 + j * 16 + l;
    const float bv = bias ? bias[col] : 0.f;     // hoisted: invariant over rows
#pragma unroll
    for (int i = 0; i < 4; ++i) {
      const int rb = rowBase + wm * 64 + i * 16 + hi * 8;
#pragma unroll
      for (int r = 0; r < 8; ++r) {
        float v = c[i][j][r] + bv;
        if (act == 1) v = fmaxf(v, 0.f);
        C[(size_t)(rb + r) * N + col] = v;
      }
    }
  }
}

// ---------------- small elementwise kernels ----------------
__device__ __forceinline__ float sigmoidf(float x) { return 1.f / (1.f + expf(-x)); }

__global__ void zero_f(float* p, size_t n) {
  size_t i = (size_t)blockIdx.x * blockDim.x + threadIdx.x;
  if (i < n) p[i] = 0.f;
}

__global__ void compute_lens(const int* __restrict__ mask, int L, int* __restrict__ lens, int N) {
  int n = blockIdx.x * blockDim.x + threadIdx.x;
  if (n >= N) return;
  int c = 0;
  for (int l = 0; l < L; ++l) c += (mask[n * L + l] == 0) ? 1 : 0;
  lens[n] = c;
}

// x[n, :] = (tok==0) ? 0 : emb[tok]; tok = tokens[n*stride + off]
__global__ void gather_emb(const int* __restrict__ tokens, int stride, int off,
                           const float* __restrict__ emb, float* __restrict__ x, int N) {
  int i = blockIdx.x * blockDim.x + threadIdx.x;
  if (i >= N * Hh) return;
  int n = i / Hh, j = i - n * Hh;
  int idx = tokens[n * stride + off];
  x[i] = (idx == 0) ? 0.f : emb[(size_t)idx * Hh + j];
}

// GRU cell elementwise, gates (r,z,n); h updated in place. Optional capture of
// h at step t == lens[n]-1 into rep (for fact/question representations).
__global__ void gru_ew(const float* __restrict__ gi, const float* __restrict__ gh,
                       const float* __restrict__ bih, const float* __restrict__ bhh,
                       float* __restrict__ h, int N,
                       const int* __restrict__ lens, int t, float* __restrict__ rep) {
  int i = blockIdx.x * blockDim.x + threadIdx.x;
  if (i >= N * Hh) return;
  int n = i / Hh, j = i - n * Hh;
  const float* gin = gi + (size_t)n * 3 * Hh;
  const float* ghn = gh + (size_t)n * 3 * Hh;
  float r  = sigmoidf(gin[j] + bih[j] + ghn[j] + bhh[j]);
  float z  = sigmoidf(gin[Hh + j] + bih[Hh + j] + ghn[Hh + j] + bhh[Hh + j]);
  float nn = tanhf(gin[2 * Hh + j] + bih[2 * Hh + j] + r * (ghn[2 * Hh + j] + bhh[2 * Hh + j]));
  float hn = (1.f - z) * nn + z * h[i];
  h[i] = hn;
  if (lens && t == lens[n] - 1) rep[i] = hn;
}

// attention-GRU step: h = g*GRU(f_t,h) + (1-g)*h, gi/gate indexed by (b*NF+f)
__global__ void att_ew(const float* __restrict__ giA, const float* __restrict__ gh,
                       const float* __restrict__ bih, const float* __restrict__ bhh,
                       const float* __restrict__ gate, float* __restrict__ h, int f) {
  int i = blockIdx.x * blockDim.x + threadIdx.x;
  if (i >= Bb * Hh) return;
  int b = i / Hh, j = i - b * Hh;
  int rowg = b * NFf + f;
  const float* gin = giA + (size_t)rowg * 3 * Hh;
  const float* ghn = gh + (size_t)b * 3 * Hh;
  float r  = sigmoidf(gin[j] + bih[j] + ghn[j] + bhh[j]);
  float z  = sigmoidf(gin[Hh + j] + bih[Hh + j] + ghn[Hh + j] + bhh[Hh + j]);
  float nn = tanhf(gin[2 * Hh + j] + bih[2 * Hh + j] + r * (ghn[2 * Hh + j] + bhh[2 * Hh + j]));
  float hn = (1.f - z) * nn + z * h[i];
  float g = gate[rowg];
  h[i] = g * hn + (1.f - g) * h[i];
}

// Z[bn, :] = [f*q, f*m, |f-q|, |f-m|]
__global__ void build_z(const float* __restrict__ frep, const float* __restrict__ q,
                        const float* __restrict__ mem, float* __restrict__ Z) {
  int i = blockIdx.x * blockDim.x + threadIdx.x;
  if (i >= Bb * NFf * Hh) return;
  int bn = i / Hh, j = i - bn * Hh;
  int b = bn / NFf;
  float f = frep[i], qq = q[(size_t)b * Hh + j], mm = mem[(size_t)b * Hh + j];
  float* zr = Z + (size_t)bn * 4 * Hh;
  zr[j]            = f * qq;
  zr[Hh + j]       = f * mm;
  zr[2 * Hh + j]   = fabsf(f - qq);
  zr[3 * Hh + j]   = fabsf(f - mm);
}

// g[row] = sigmoid(dot(Hg[row,:], w2) + b2); one wave32 per row
__global__ void gate_dot(const float* __restrict__ Hg, const float* __restrict__ w2,
                         const float* __restrict__ b2, float* __restrict__ g, int N) {
  int wave = (blockIdx.x * blockDim.x + threadIdx.x) >> 5;
  int lane = threadIdx.x & 31;
  if (wave >= N) return;
  float s = 0.f;
  for (int j = lane; j < Hh; j += 32) s += Hg[(size_t)wave * Hh + j] * w2[j];
  for (int o = 16; o > 0; o >>= 1) s += __shfl_down(s, o, 32);
  if (lane == 0) g[wave] = sigmoidf(s + b2[0]);
}

// ans_in[b, :] = [emb[START_IDX,:], q_rep[b,:]]
__global__ void build_ansin(const float* __restrict__ emb, const float* __restrict__ q,
                            float* __restrict__ ai) {
  int i = blockIdx.x * blockDim.x + threadIdx.x;
  if (i >= Bb * 2 * Hh) return;
  int b = i / (2 * Hh), j = i - b * 2 * Hh;
  ai[i] = (j < Hh) ? emb[(size_t)START_IDX * Hh + j] : q[(size_t)b * Hh + (j - Hh)];
}

// per-row log_softmax over V, write to out row (b*ND + d)
__global__ void logsoftmax_row(const float* __restrict__ logits, float* __restrict__ out, int d) {
  __shared__ float red[256];
  int b = blockIdx.x;
  const float* x = logits + (size_t)b * Vv;
  float* o = out + ((size_t)b * NDd + d) * Vv;
  int tid = threadIdx.x;
  float m = -INFINITY;
  for (int v = tid; v < Vv; v += 256) m = fmaxf(m, x[v]);
  red[tid] = m; __syncthreads();
  for (int s = 128; s > 0; s >>= 1) { if (tid < s) red[tid] = fmaxf(red[tid], red[tid + s]); __syncthreads(); }
  m = red[0]; __syncthreads();
  float sum = 0.f;
  for (int v = tid; v < Vv; v += 256) sum += expf(x[v] - m);
  red[tid] = sum; __syncthreads();
  for (int s = 128; s > 0; s >>= 1) { if (tid < s) red[tid] += red[tid + s]; __syncthreads(); }
  float lse = m + logf(red[0]);
  for (int v = tid; v < Vv; v += 256) o[v] = x[v] - lse;
}

// ---------------- host orchestration ----------------
static inline int cdiv(int a, int b) { return (a + b - 1) / b; }

extern "C" void kernel_launch(void* const* d_in, const int* in_sizes, int n_in,
                              void* d_out, int out_size, void* d_ws, size_t ws_size,
                              hipStream_t stream) {
  (void)in_sizes; (void)n_in; (void)out_size; (void)ws_size;

  const int*   facts   = (const int*)d_in[0];
  const int*   fmask   = (const int*)d_in[1];
  const int*   quest   = (const int*)d_in[2];
  const int*   qmask   = (const int*)d_in[3];
  // d_in[4] = num_decode (device scalar); compile-time NDd==4 per reference
  const float* emb     = (const float*)d_in[5];
  const float* ig_Wih  = (const float*)d_in[6];
  const float* ig_Whh  = (const float*)d_in[7];
  const float* ig_bih  = (const float*)d_in[8];
  const float* ig_bhh  = (const float*)d_in[9];
  const float* qg_Wih  = (const float*)d_in[10];
  const float* qg_Whh  = (const float*)d_in[11];
  const float* qg_bih  = (const float*)d_in[12];
  const float* qg_bhh  = (const float*)d_in[13];
  const float* g_w1    = (const float*)d_in[14];
  const float* g_b1    = (const float*)d_in[15];
  const float* g_w2    = (const float*)d_in[16];
  const float* g_b2    = (const float*)d_in[17];
  const float* a_Wih   = (const float*)d_in[18];
  const float* a_Whh   = (const float*)d_in[19];
  const float* a_bih   = (const float*)d_in[20];
  const float* a_bhh   = (const float*)d_in[21];
  const float* m_Wih   = (const float*)d_in[22];
  const float* m_Whh   = (const float*)d_in[23];
  const float* m_bih   = (const float*)d_in[24];
  const float* m_bhh   = (const float*)d_in[25];
  const float* ansWih  = (const float*)d_in[26];
  const float* ansWhh  = (const float*)d_in[27];
  const float* ansbih  = (const float*)d_in[28];
  const float* ansbhh  = (const float*)d_in[29];
  const float* fc_w    = (const float*)d_in[30];
  const float* fc_b    = (const float*)d_in[31];
  float* out = (float*)d_out;

  const int NFb = Bb * NFf;         // 5120
  // ---- workspace layout (floats), regions reused across phases (~107MB) ----
  float* ws = (float*)d_ws;
  size_t o = 0;
  float* giBuf  = ws + o; o += (size_t)NFb * 3 * Hh;     // 7.86M: per-step gi; later gi_att
  float* ghBuf  = ws + o; o += (size_t)NFb * 4 * Hh;     // 10.49M: gh / Z / logits
  float* xbBuf  = ws + o; o += (size_t)NFb * Hh;         // 2.62M: embedded x_t / Hg
  float* hBuf   = ws + o; o += (size_t)NFb * Hh;         // 2.62M: recurrent hidden / e
  float* frep   = ws + o; o += (size_t)NFb * Hh;         // 2.62M: fact representations
  float* qrep   = ws + o; o += (size_t)Bb * Hh;
  float* mem    = ws + o; o += (size_t)Bb * Hh;
  float* gate   = ws + o; o += (size_t)NFb;
  float* ansin  = ws + o; o += (size_t)Bb * 2 * Hh;
  float* giS    = ws + o; o += (size_t)Bb * 3 * Hh;      // small gi (memory/answer GRUs)
  int*   flens  = (int*)(ws + o); o += NFb;
  int*   qlens  = (int*)(ws + o); o += Bb;

  auto gemm = [&](const float* A, const float* W, const float* bias, float* C,
                  int M, int N, int K, int act) {
    dim3 grid(N / TN, M / TM);
    wmma_gemm_nt<<<grid, 256, 0, stream>>>(A, W, bias, C, M, N, K, act);
  };
  auto zero = [&](float* p, size_t n) {
    zero_f<<<(int)((n + 255) / 256), 256, 0, stream>>>(p, n);
  };
  const int EW_F = cdiv(NFb * Hh, 256);   // 5120*512 elementwise grid
  const int EW_B = cdiv(Bb * Hh, 256);    // 128*512 elementwise grid

  // ================= input module: GRU over facts =================
  compute_lens<<<cdiv(NFb, 256), 256, 0, stream>>>(fmask, Ll, flens, NFb);
  zero(hBuf, (size_t)NFb * Hh);
  for (int t = 0; t < Ll; ++t) {
    gather_emb<<<EW_F, 256, 0, stream>>>(facts, Ll, t, emb, xbBuf, NFb);
    gemm(xbBuf, ig_Wih, nullptr, giBuf, NFb, 3 * Hh, Hh, 0);
    gemm(hBuf,  ig_Whh, nullptr, ghBuf, NFb, 3 * Hh, Hh, 0);
    gru_ew<<<EW_F, 256, 0, stream>>>(giBuf, ghBuf, ig_bih, ig_bhh, hBuf, NFb, flens, t, frep);
  }

  // ================= question module =================
  compute_lens<<<cdiv(Bb, 256), 256, 0, stream>>>(qmask, QLl, qlens, Bb);
  zero(hBuf, (size_t)Bb * Hh);
  for (int t = 0; t < QLl; ++t) {
    gather_emb<<<EW_B, 256, 0, stream>>>(quest, QLl, t, emb, xbBuf, Bb);
    gemm(xbBuf, qg_Wih, nullptr, giBuf, Bb, 3 * Hh, Hh, 0);
    gemm(hBuf,  qg_Whh, nullptr, ghBuf, Bb, 3 * Hh, Hh, 0);
    gru_ew<<<EW_B, 256, 0, stream>>>(giBuf, ghBuf, qg_bih, qg_bhh, hBuf, Bb, qlens, t, qrep);
  }

  // ================= episodic memory =================
  hipMemcpyAsync(mem, qrep, (size_t)Bb * Hh * sizeof(float), hipMemcpyDeviceToDevice, stream);
  // attention-GRU gi depends only on facts: precompute once (5120 x 1536)
  gemm(frep, a_Wih, nullptr, giBuf, NFb, 3 * Hh, Hh, 0);
  for (int ep = 0; ep < EPISODES; ++ep) {
    // all gates for this episode in one shot
    build_z<<<EW_F, 256, 0, stream>>>(frep, qrep, mem, ghBuf);                 // Z: 5120x2048
    gemm(ghBuf, g_w1, g_b1, xbBuf, NFb, Hh, 4 * Hh, 1);                        // Hg = relu(Z@w1+b1)
    gate_dot<<<NFb / 8, 256, 0, stream>>>(xbBuf, g_w2, g_b2, gate, NFb);       // g = sigm(Hg@w2+b2)
    zero(hBuf, (size_t)Bb * Hh);
    for (int f = 0; f < NFf; ++f) {
      gemm(hBuf, a_Whh, nullptr, ghBuf, Bb, 3 * Hh, Hh, 0);
      att_ew<<<EW_B, 256, 0, stream>>>(giBuf, ghBuf, a_bih, a_bhh, gate, hBuf, f);
    }
    // memory = GRUCell(e=hBuf, memory)
    gemm(hBuf, m_Wih, nullptr, giS,   Bb, 3 * Hh, Hh, 0);
    gemm(mem,  m_Whh, nullptr, ghBuf, Bb, 3 * Hh, Hh, 0);
    gru_ew<<<EW_B, 256, 0, stream>>>(giS, ghBuf, m_bih, m_bhh, mem, Bb, nullptr, 0, nullptr);
  }

  // ================= answer module =================
  build_ansin<<<cdiv(Bb * 2 * Hh, 256), 256, 0, stream>>>(emb, qrep, ansin);
  gemm(ansin, ansWih, nullptr, giS, Bb, 3 * Hh, 2 * Hh, 0);   // fixed input -> gi once
  for (int d = 0; d < NDd; ++d) {
    gemm(mem, ansWhh, nullptr, ghBuf, Bb, 3 * Hh, Hh, 0);
    gru_ew<<<EW_B, 256, 0, stream>>>(giS, ghBuf, ansbih, ansbhh, mem, Bb, nullptr, 0, nullptr);
    gemm(mem, fc_w, fc_b, ghBuf, Bb, Vv, Hh, 0);              // logits: 128 x 32000
    logsoftmax_row<<<Bb, 256, 0, stream>>>(ghBuf, out, d);
  }
}
